// AE10_22832046145592
// MI455X (gfx1250) — matrix operand order repaired and checked by reading, not compiled, not run on hardware
//
#include <hip/hip_runtime.h>
#include <hip/hip_bf16.h>
#include <hip/hip_fp16.h>

typedef __attribute__((ext_vector_type(16))) _Float16 v16h;
typedef __attribute__((ext_vector_type(8)))  float    v8f;

#define B_    16
#define CIN   3
#define H_    224
#define W_    224
#define COUT  128
#define KTOT  147      // 3 * 7 * 7
#define KPAD  160      // 5 * 32 (WMMA K-steps)
#define KB_N  5        // K blocks of 32
#define PIX   (H_ * W_)   // 50176
#define TILE_N 64      // pixels per workgroup
#define NT     4       // 16-wide N tiles per workgroup
#define EPS    1e-5f
#define NEG_INF (-3.4e38f)
#define WSWZ_HALVES (8 * KB_N * 32 * 16)   // 20480 f16 = 40960 B

// ---------------------------------------------------------------------------
// Kernel 0: convert conv weights to f16 and swizzle into the exact per-lane
// WMMA A-fragment layout: wswz[wid][kb][lane][e], so kernel 1 fetches one
// 32-byte contiguous v16h per (wave, K-block) straight from L2.
// A-fragment mapping used here (16-bit A 16x32, contiguous K, same order as B):
//   M = wid*16 + (lane&15)
//   K = kb*32 + (lane>=16 ? 16 : 0) + e
// ---------------------------------------------------------------------------
__global__ __launch_bounds__(256) void AE10_wswz(
    const float* __restrict__ conv_w,   // [128,147]
    _Float16* __restrict__ wswz)        // [8][5][32][16] halves
{
    const int i = blockIdx.x * 256 + threadIdx.x;   // 0 .. 20479
    if (i >= WSWZ_HALVES) return;
    const int e    = i & 15;
    const int lane = (i >> 4) & 31;
    const int kb   = (i >> 9) % KB_N;
    const int wid  = i / (KB_N * 32 * 16);
    const int m = wid * 16 + (lane & 15);
    const int k = kb * 32 + ((lane >> 4) ? 16 : 0) + e;
    wswz[i] = (k < KTOT) ? (_Float16)conv_w[m * KTOT + k] : (_Float16)0.f;
}

// ---------------------------------------------------------------------------
// Kernel 1: fused implicit-GEMM conv (WMMA f16->f32) + bias + BN + ReLU +
// per-pixel top-3 over channels. One WG = 8 waves = 128 channels x 64 pixels.
// ---------------------------------------------------------------------------
__global__ __launch_bounds__(256) void AE10_conv_top3(
    const float* __restrict__ x,        // [16,3,224,224]
    const _Float16* __restrict__ wswz,  // pre-swizzled A fragments (L2-resident)
    const float* __restrict__ conv_b,
    const float* __restrict__ gamma,
    const float* __restrict__ beta,
    const float* __restrict__ mean,
    const float* __restrict__ var,
    float* __restrict__ top3val,        // [16,3,PIX]
    int*   __restrict__ top3idx,        // [16,3,PIX]
    float* __restrict__ maxcopy)        // [16,PIX]
{
    __shared__ _Float16 lds_p[TILE_N * KPAD];   // 20480 B im2col patch (f16)
    __shared__ float    lds_y[COUT * TILE_N];   // 32768 B conv+BN+ReLU tile
    __shared__ float    lds_scale[COUT];        //   512 B
    __shared__ float    lds_shift[COUT];        //   512 B  => 54272 B total

    const int t  = threadIdx.x;
    const int b  = blockIdx.y;
    const int p0 = blockIdx.x * TILE_N;         // 784 tiles per image

    const int lane   = t & 31;
    const int wid    = t >> 5;
    const int laneHi = lane >> 4;
    const int ll     = lane & 15;

    // ---- preload all 5 A fragments (32B vector loads, overlaps staging) ----
    v16h a[KB_N];
#pragma unroll
    for (int kb = 0; kb < KB_N; ++kb)
        a[kb] = *(const v16h*)(wswz + ((size_t)(wid * KB_N + kb) * 32 + lane) * 16);

    // ---- stage im2col patch: branchless clamp + select, pipelined loads ----
#pragma unroll 4
    for (int e = t; e < TILE_N * KPAD; e += 256) {
        const int col = e / KPAD;
        const int k   = e - col * KPAD;
        const int p   = p0 + col;
        const int h   = p / W_;
        const int w   = p - h * W_;
        const int kk  = (k < KTOT) ? k : 0;
        const int ci  = kk / 49;
        const int r   = kk - ci * 49;
        const int kh  = r / 7;
        const int kw  = r - kh * 7;
        const int hi  = h + kh - 3;
        const int wi  = w + kw - 3;
        const bool inb = (k < KTOT) & (hi >= 0) & (hi < H_) & (wi >= 0) & (wi < W_);
        const int hic = min(max(hi, 0), H_ - 1);
        const int wic = min(max(wi, 0), W_ - 1);
        const float v = x[(((size_t)b * CIN + ci) * H_ + hic) * W_ + wic];
        lds_p[e] = (_Float16)(inb ? v : 0.f);
    }
    if (t < COUT) {
        const float s = gamma[t] * rsqrtf(var[t] + EPS);
        lds_scale[t] = s;
        lds_shift[t] = (conv_b[t] - mean[t]) * s + beta[t];
    }
    __syncthreads();

    const v8f zero8 = {0.f, 0.f, 0.f, 0.f, 0.f, 0.f, 0.f, 0.f};
    v8f c[NT];
#pragma unroll
    for (int n = 0; n < NT; ++n) c[n] = zero8;

    // ---- K loop: 5 x (A in regs, 4x ds_load_b128 pair + WMMA) ----
#pragma unroll
    for (int kb = 0; kb < KB_N; ++kb) {
#pragma unroll
        for (int n = 0; n < NT; ++n) {
            // B fragment (16-bit B 32x16): lanes 0-15 col=lane K 0-15,
            // lanes 16-31 col=lane-16 K 16-31 — 16 contiguous halves.
            v16h bm;
            const int bbase = (n * 16 + ll) * KPAD + kb * 32 + laneHi * 16;
#pragma unroll
            for (int e = 0; e < 16; ++e) bm[e] = lds_p[bbase + e];

            c[n] = __builtin_amdgcn_wmma_f32_16x16x32_f16(
                       false, a[kb], false, bm, (short)0, c[n], false, false);
        }
    }

    // ---- epilogue: BN + ReLU, write 128x64 tile to LDS ----
#pragma unroll
    for (int n = 0; n < NT; ++n) {
        const int col = n * 16 + ll;
#pragma unroll
        for (int r = 0; r < 8; ++r) {
            const int m = wid * 16 + r + laneHi * 8;
            float y = c[n][r] * lds_scale[m] + lds_shift[m];
            y = fmaxf(y, 0.f);
            lds_y[m * TILE_N + col] = y;
        }
    }
    __syncthreads();

    // ---- per-pixel top-3 over 128 channels (tie -> lower channel index) ----
    if (t < TILE_N) {
        float v0 = NEG_INF, v1 = NEG_INF, v2 = NEG_INF;
        int   i0 = 0, i1 = 0, i2 = 0;
        for (int m = 0; m < COUT; ++m) {
            const float v = lds_y[m * TILE_N + t];
            if (v > v0)      { v2 = v1; i2 = i1; v1 = v0; i1 = i0; v0 = v; i0 = m; }
            else if (v > v1) { v2 = v1; i2 = i1; v1 = v;  i1 = m; }
            else if (v > v2) { v2 = v;  i2 = m; }
        }
        const int p = p0 + t;
        const size_t base = (size_t)b * 3 * PIX;
        top3val[base + 0 * PIX + p] = v0;
        top3val[base + 1 * PIX + p] = v1;
        top3val[base + 2 * PIX + p] = v2;
        top3idx[base + 0 * PIX + p] = i0;
        top3idx[base + 1 * PIX + p] = i1;
        top3idx[base + 2 * PIX + p] = i2;
        maxcopy[(size_t)b * PIX + p] = v0;
    }
}

// ---------------------------------------------------------------------------
// Kernel 2: per-image top-128 spatial selection (value desc, index asc) over
// the channel-max map, then gather of top-3 values / indices at those pixels.
// ---------------------------------------------------------------------------
__global__ __launch_bounds__(256) void AE10_topk_gather(
    const float* __restrict__ top3val,
    const int*   __restrict__ top3idx,
    float* __restrict__ maxcopy,
    float* __restrict__ out)
{
    __shared__ float lds_v[256];
    __shared__ int   lds_i[256];
    __shared__ int   sel[128];

    const int t = threadIdx.x;
    const int b = blockIdx.x;
    float* src = maxcopy + (size_t)b * PIX;

    for (int j = 0; j < 128; ++j) {
        float best = NEG_INF;
        int   bidx = 0x7fffffff;
#pragma unroll 4
        for (int p = t; p < PIX; p += 256) {
            const float v = src[p];
            if (v > best || (v == best && p < bidx)) { best = v; bidx = p; }
        }
        lds_v[t] = best;
        lds_i[t] = bidx;
        __syncthreads();
        for (int s = 128; s > 0; s >>= 1) {
            if (t < s) {
                const float ov = lds_v[t + s];
                const int   oi = lds_i[t + s];
                if (ov > lds_v[t] || (ov == lds_v[t] && oi < lds_i[t])) {
                    lds_v[t] = ov; lds_i[t] = oi;
                }
            }
            __syncthreads();
        }
        if (t == 0) {
            sel[j] = lds_i[0];
            src[lds_i[0]] = NEG_INF;   // knock out for next round
        }
        __threadfence();
        __syncthreads();
    }

    // gather: out = [idx3: 16x3x128][top3: 16x3x128][idx: 16x128], all f32
    if (t < 128) {
        const int p = sel[t];
        float* out_idx3 = out;
        float* out_val3 = out + 16 * 3 * 128;
        float* out_sidx = out + 2 * 16 * 3 * 128;
#pragma unroll
        for (int r = 0; r < 3; ++r) {
            const size_t src_off = ((size_t)b * 3 + r) * PIX + p;
            out_val3[(b * 3 + r) * 128 + t] = top3val[src_off];
            out_idx3[(b * 3 + r) * 128 + t] = (float)top3idx[src_off];
        }
        out_sidx[b * 128 + t] = (float)p;
    }
}

extern "C" void kernel_launch(void* const* d_in, const int* in_sizes, int n_in,
                              void* d_out, int out_size, void* d_ws, size_t ws_size,
                              hipStream_t stream) {
    const float* x      = (const float*)d_in[0];
    const float* conv_w = (const float*)d_in[1];
    const float* conv_b = (const float*)d_in[2];
    const float* gamma  = (const float*)d_in[3];
    const float* beta   = (const float*)d_in[4];
    const float* mean   = (const float*)d_in[5];
    const float* var    = (const float*)d_in[6];

    // ws: wswz (40960 B) | top3val (16*3*PIX f32) | top3idx (16*3*PIX i32)
    //   | maxcopy (16*PIX f32)   ~= 22.6 MB total
    _Float16* wswz   = (_Float16*)d_ws;
    float*   top3val = (float*)((char*)d_ws + WSWZ_HALVES * sizeof(_Float16));
    int*     top3idx = (int*)(top3val + (size_t)B_ * 3 * PIX);
    float*   maxcopy = (float*)(top3idx + (size_t)B_ * 3 * PIX);

    AE10_wswz<<<(WSWZ_HALVES + 255) / 256, 256, 0, stream>>>(conv_w, wswz);

    dim3 grid1(PIX / TILE_N, B_);     // 784 x 16 workgroups
    AE10_conv_top3<<<grid1, 256, 0, stream>>>(
        x, wswz, conv_b, gamma, beta, mean, var,
        top3val, top3idx, maxcopy);

    AE10_topk_gather<<<B_, 256, 0, stream>>>(
        top3val, top3idx, maxcopy, (float*)d_out);
}
